// Allegro_54674933678516
// MI455X (gfx1250) — matrix operand order repaired and checked by reading, not compiled, not run on hardware
//
#include <hip/hip_runtime.h>
#include <hip/hip_bf16.h>
#include <math.h>

#define EDGES  32768
#define NNODES 2048

typedef __attribute__((ext_vector_type(16))) _Float16 v16h;
typedef __attribute__((ext_vector_type(8)))  _Float16 v8h;
typedef __attribute__((ext_vector_type(4)))  _Float16 v4h;
typedef __attribute__((ext_vector_type(8)))  float    v8f;

// ---------------- CG path tables (l3-grouped, python iteration order) ----------
__constant__ int P_L1[34]  = {0,1,2,3, 0,1,1,1,2,2,2,3,3, 0,1,1,1,2,2,2,2,3,3,3, 0,1,1,2,2,2,3,3,3,3};
__constant__ int P_L2[34]  = {0,1,2,3, 1,0,1,2,1,2,3,2,3, 2,1,2,3,0,1,2,3,1,2,3, 3,2,3,1,2,3,0,1,2,3};
__constant__ int P_TOFF[34]= {0,1,10,35,
                              84,93,102,129,174,219,294,399,504,
                              651,676,721,796,901,926,1001,1126,1301,1406,1581,
                              1826,1875,1980,2127,2232,2407,2652,2701,2848,3093};
__constant__ int P_G0[5]   = {0,4,13,24,34};
__constant__ int SLO[4]    = {0,1,4,9};

// ---------------- device CG / real-coupling construction ----------------------
__device__ double dfact(int n){ double r=1.0; for(int i=2;i<=n;++i) r*=(double)i; return r; }

__device__ double cgc(int j1,int m1,int j2,int m2,int j3,int m3){
  if (m1+m2 != m3) return 0.0;
  double pre = sqrt((2.0*j3+1.0)*dfact(j1+j2-j3)*dfact(j1-j2+j3)*dfact(-j1+j2+j3)/dfact(j1+j2+j3+1));
  pre *= sqrt(dfact(j3+m3)*dfact(j3-m3)*dfact(j1-m1)*dfact(j1+m1)*dfact(j2-m2)*dfact(j2+m2));
  int kmin = 0;
  if (j2-j3-m1 > kmin) kmin = j2-j3-m1;
  if (j1-j3+m2 > kmin) kmin = j1-j3+m2;
  int kmax = j1+j2-j3;
  if (j1-m1 < kmax) kmax = j1-m1;
  if (j2+m2 < kmax) kmax = j2+m2;
  double s = 0.0;
  for (int k=kmin; k<=kmax; ++k){
    double den = dfact(k)*dfact(j1+j2-j3-k)*dfact(j1-m1-k)*dfact(j2+m2-k)*dfact(j3-j2+m1+k)*dfact(j3-j1-m2+k);
    s += ((k&1)? -1.0 : 1.0)/den;
  }
  return pre*s;
}

__device__ void umat(int l, double* Ur, double* Ui){
  int n = 2*l+1;
  for (int i=0;i<n*n;++i){ Ur[i]=0.0; Ui[i]=0.0; }
  Ur[l*n+l] = 1.0;
  double s2 = 1.0/sqrt(2.0);
  for (int m=1;m<=l;++m){
    double sgn = (m&1)? -1.0 : 1.0;
    Ur[(l+m)*n + (l-m)] = s2;
    Ur[(l+m)*n + (l+m)] = sgn*s2;
    Ui[(l-m)*n + (l-m)] = s2;
    Ui[(l-m)*n + (l+m)] = -sgn*s2;
  }
}

__global__ void cg_init_kernel(float* __restrict__ Tg){
  __shared__ double Trs[343];
  __shared__ double Tis[343];
  __shared__ double red[64];
  const int g  = blockIdx.x;
  const int l3 = (g<4)?0:(g<13)?1:(g<24)?2:3;
  const int l1 = P_L1[g], l2 = P_L2[g];
  const int n1 = 2*l1+1, n2 = 2*l2+1, n3 = 2*l3+1;
  double U1r[49],U1i[49],U2r[49],U2i[49],U3r[49],U3i[49];
  umat(l1,U1r,U1i); umat(l2,U2r,U2i); umat(l3,U3r,U3i);
  const int tot = n1*n2*n3;
  for (int idx=threadIdx.x; idx<tot; idx+=blockDim.x){
    int c = idx % n3; int ab = idx / n3; int b = ab % n2; int a = ab / n2;
    double ar=0.0, ai=0.0;
    for (int mm=0; mm<n1; ++mm){
      for (int nn=0; nn<n2; ++nn){
        int kk = mm+nn-l1-l2+l3;
        if (kk<0 || kk>=n3) continue;
        double cv = cgc(l1,mm-l1,l2,nn-l2,l3,kk-l3);
        if (cv==0.0) continue;
        double x1=U1r[a*n1+mm], y1=U1i[a*n1+mm];
        double x2=U2r[b*n2+nn], y2=U2i[b*n2+nn];
        double zr = x1*x2 - y1*y2, zi = x1*y2 + y1*x2;
        double x3=U3r[c*n3+kk], y3=U3i[c*n3+kk];
        ar += (zr*x3 + zi*y3)*cv;
        ai += (zi*x3 - zr*y3)*cv;
      }
    }
    Trs[idx]=ar; Tis[idx]=ai;
  }
  __syncthreads();
  double pr=0.0, pi=0.0;
  for (int idx=threadIdx.x; idx<tot; idx+=blockDim.x){ pr+=Trs[idx]*Trs[idx]; pi+=Tis[idx]*Tis[idx]; }
  red[threadIdx.x]=pr; __syncthreads();
  if (threadIdx.x==0){ double s=0; for(int i=0;i<64;++i) s+=red[i]; red[0]=s; }
  __syncthreads();
  double nr2 = red[0];
  __syncthreads();
  red[threadIdx.x]=pi; __syncthreads();
  if (threadIdx.x==0){ double s=0; for(int i=0;i<64;++i) s+=red[i]; red[0]=s; }
  __syncthreads();
  double ni2 = red[0];
  bool useR = (nr2 >= ni2);
  double nrm = sqrt(useR? nr2 : ni2);
  for (int idx=threadIdx.x; idx<tot; idx+=blockDim.x)
    Tg[P_TOFF[g]+idx] = (float)((useR? Trs[idx] : Tis[idx]) / nrm);
}

// ---------------- weight prep: f32 W[K][N] -> f16 Wt[N][KP] (zero-padded) -----
__global__ void wprep_kernel(const float* __restrict__ W, _Float16* __restrict__ Wt,
                             int K, int N, int KP){
  int idx = blockIdx.x*blockDim.x + threadIdx.x;
  if (idx >= N*KP) return;
  int n = idx / KP, k = idx % KP;
  Wt[(size_t)n*KP + k] = (k < K) ? (_Float16)W[(size_t)k*N + n] : (_Float16)0.f;
}

// ---------------- edge geometry + embedding gather ----------------------------
__global__ void edge_init_kernel(const float* __restrict__ vec, const float* __restrict__ emb,
                                 const int* __restrict__ senders, const int* __restrict__ receivers,
                                 const int* __restrict__ species,
                                 float* __restrict__ envb, float* __restrict__ Yb, float* __restrict__ x0){
  int e = blockIdx.x*blockDim.x + threadIdx.x;
  if (e >= EDGES) return;
  float vx=vec[3*e], vy=vec[3*e+1], vz=vec[3*e+2];
  float d  = sqrtf(vx*vx+vy*vy+vz*vz);
  float dd = (d==0.f)? 1.f : d;
  float x=vx/dd, y=vy/dd, z=vz/dd;
  float d2=d*d, d3=d2*d, d6=d3*d3, d7=d6*d, d8=d7*d;
  float env = (d<1.f)? (1.f - 28.f*d6 + 48.f*d7 - 21.f*d8) : 0.f;
  envb[e] = env;
  float mask = (d==0.f)? 0.f : 1.f;
  const float PI_ = 3.14159265358979323846f;
  for (int n=1; n<=8; ++n)
    x0[(size_t)e*72 + n-1] = mask * 1.41421356237f * sinf((float)n*PI_*d)/dd * env;
  int ss = species[senders[e]], sr = species[receivers[e]];
  for (int j=0;j<32;++j){
    x0[(size_t)e*72 + 8  + j] = mask*emb[ss*32+j];
    x0[(size_t)e*72 + 40 + j] = mask*emb[sr*32+j];
  }
  float x2=x*x, y2=y*y, z2=z*z;
  const float s3=1.7320508075688772f, s15=3.872983346207417f, s5=2.23606797749979f;
  const float s358=2.0916500663351889f, s105=10.246950765959598f, s218=1.6201851746019651f, s7=2.6457513110645907f;
  float* Ye = Yb + (size_t)e*16;
  Ye[0]=1.f;              Ye[1]=s3*y;                 Ye[2]=s3*z;                 Ye[3]=s3*x;
  Ye[4]=s15*x*y;          Ye[5]=s15*y*z;              Ye[6]=0.5f*s5*(3.f*z2-1.f); Ye[7]=s15*x*z;
  Ye[8]=0.5f*s15*(x2-y2); Ye[9]=s358*y*(3.f*x2-y2);   Ye[10]=s105*x*y*z;          Ye[11]=s218*y*(5.f*z2-1.f);
  Ye[12]=0.5f*s7*(5.f*z2-3.f)*z; Ye[13]=s218*x*(5.f*z2-1.f); Ye[14]=0.5f*s105*(x2-y2)*z; Ye[15]=s358*x*(x2-y2);
}

// ---------------- WMMA GEMM: C = post(A[E,K] @ W[K,N] / sqrt(K)) --------------
// Wt is pre-transposed padded f16 [N][KP].
// MODE 0: plain   1: silu   2: env*y   3: (Xold + a^2*env*y)/(1+a^2)
constexpr int rt_for(int K, int N){
  int KP = ((K+31)/32)*32, NT = N/16;
  int RT0 = (NT >= 8) ? 1 : (8/NT);
  int LIM = 32768/(16*KP*2); if (LIM < 1) LIM = 1;
  return RT0 < LIM ? RT0 : LIM;
}

template<int K, int N, int MODE>
__global__ void __launch_bounds__(256)
gemm_wmma(const float* __restrict__ A, const _Float16* __restrict__ Wt,
          float* __restrict__ C, const float* __restrict__ envb,
          const float* __restrict__ Xold, const float* __restrict__ alpha_p){
  constexpr int KP = ((K+31)/32)*32;
  constexpr int NT = N/16;
  constexpr int RT = rt_for(K, N);
  constexpr int MR = 16*RT;
  static_assert((K & 7) == 0, "K must be a multiple of 8 for vectorized staging");
  __shared__ __align__(32) _Float16 As[MR*KP];
  const int m0  = blockIdx.x * MR;
  const int tid = threadIdx.x;
  // stage A tile: float4 global loads -> packed f16 -> 8-byte LDS stores
  constexpr int KC = K/4;
  for (int i = tid; i < MR*KC; i += 256){
    int r = i / KC, kc = (i % KC)*4;
    const float4 v = *(const float4*)&A[(size_t)(m0 + r)*K + kc];
    v4h hv; hv[0]=(_Float16)v.x; hv[1]=(_Float16)v.y; hv[2]=(_Float16)v.z; hv[3]=(_Float16)v.w;
    *(v4h*)&As[r*KP + kc] = hv;
  }
  if (KP > K){
    for (int i = tid; i < MR*(KP-K); i += 256){
      int r = i / (KP-K), k = K + (i % (KP-K));
      As[r*KP + k] = (_Float16)0.f;
    }
  }
  __syncthreads();

  const int wave = tid >> 5;
  const int lane = tid & 31;
  const int half = lane >> 4;
  const int lm   = lane & 15;
  for (int tt = wave; tt < RT*NT; tt += 8){
    const int mt = tt / NT, nt = tt % NT;
    const int rb = mt*16;          // row tile base inside block
    const int n0 = nt*16;
    const _Float16* wp = Wt + (size_t)(n0 + lm)*KP;
    __builtin_prefetch(wp, 0, 1);
    v8f acc = {};
    for (int k0 = 0; k0 < KP; k0 += 32){
      // A fragment: 16 contiguous halves per lane (k = 16*half + s, matches B)
      v16h a  = *(const v16h*)&As[(rb + lm)*KP + k0 + half*16];
      // B fragment: 16 contiguous halves in transposed weight (32x16 f16 B layout)
      v16h b  = *(const v16h*)&wp[k0 + half*16];
      acc = __builtin_amdgcn_wmma_f32_16x16x32_f16(false, a, false, b, (short)0, acc, false, false);
    }
    const float scale = rsqrtf((float)K);
    #pragma unroll
    for (int r = 0; r < 8; ++r){
      int row = m0 + rb + r + 8*half;   // D: VGPR r -> row r+8*half, col lane%16
      int col = n0 + lm;
      float g = acc[r] * scale;
      if (MODE == 1){ g = g * __builtin_amdgcn_rcpf(1.f + __expf(-g)); }
      else if (MODE == 2){ g = envb[row] * g; }
      else if (MODE == 3){
        float a2 = alpha_p[0]; a2 *= a2;
        g = (Xold[(size_t)row*N + col] + a2 * envb[row] * g) * __builtin_amdgcn_rcpf(1.f + a2);
      }
      C[(size_t)row*N + col] = g;
    }
  }
}

// ---------------- misc kernels -------------------------------------------------
__global__ void zero_kernel(float* __restrict__ p, int n){
  int i = blockIdx.x*blockDim.x + threadIdx.x;
  if (i < n) p[i] = 0.f;
}

__global__ void scatter_kernel(const float* __restrict__ w, const float* __restrict__ Yb,
                               const int* __restrict__ senders, float* __restrict__ node){
  int idx = blockIdx.x*blockDim.x + threadIdx.x;
  if (idx >= EDGES*256) return;
  int e = idx >> 8; int r = idx & 255; int m = r >> 4; int k = r & 15;
  float val = w[e*16+m] * Yb[e*16+k];
  atomicAdd(&node[(size_t)senders[e]*256 + r], val);
}

// ---------------- full tensor product (layer 1) + Vn dense + xc concat --------
__global__ void __launch_bounds__(128)
tp1_kernel(const float* __restrict__ node, const float* __restrict__ w0, const float* __restrict__ Yb,
           const float* __restrict__ x, const float* __restrict__ Tg, const int* __restrict__ senders,
           const float* __restrict__ vareps,
           const float* __restrict__ Wv1, const float* __restrict__ Wv2, const float* __restrict__ Wv3,
           float* __restrict__ xc, float* __restrict__ Vn){
  __shared__ float sm[3008];
  float* As = sm;          // [16][16] wY
  float* Bs = sm + 256;    // [16][16] V
  float* tp = sm + 512;    // tp0@0(64), tp1@64(432), tp2@496(880), tp3@1376(1120)
  const int e   = blockIdx.x;
  const int tid = threadIdx.x;
  const int s   = senders[e];
  float ve  = vareps[0];
  float eps = rsqrtf(1.f + ve*ve);
  for (int i = tid; i < 256; i += 128){
    As[i] = eps * node[(size_t)s*256 + i];
    Bs[i] = w0[e*16 + (i>>4)] * Yb[(size_t)e*16 + (i&15)];
  }
  __syncthreads();
  const int l3base[4] = {0, 64, 496, 1376};
  for (int g = 0; g < 34; ++g){
    int l3 = (g<4)?0:(g<13)?1:(g<24)?2:3;
    int l1 = P_L1[g], l2 = P_L2[g];
    int n1 = 2*l1+1, n2 = 2*l2+1, nk = 2*l3+1;
    int o1 = SLO[l1], o2 = SLO[l2];
    int toff = P_TOFF[g];
    float scale = sqrtf((float)(2*l3+1));
    int tot = nk*16;
    for (int ii = tid; ii < tot; ii += 128){
      int k = ii >> 4; int m = ii & 15;
      float acc = 0.f;
      for (int a = 0; a < n1; ++a){
        float av = As[m*16 + o1 + a];
        for (int b = 0; b < n2; ++b)
          acc += Tg[toff + (a*n2+b)*nk + k] * av * Bs[m*16 + o2 + b];
      }
      int chan = (g - P_G0[l3])*16 + m;
      tp[l3base[l3] + chan*nk + k] = acc * scale;
    }
  }
  __syncthreads();
  for (int j = tid; j < 320; j += 128)
    xc[(size_t)e*320 + j] = (j < 256) ? x[(size_t)e*256 + j] : tp[j - 256];
  for (int idx = tid; idx < 256; idx += 128){
    int dch = idx >> 4, kk = idx & 15;
    float out = 0.f;
    if (kk > 0){
      const float* Wp; const float* tb; int nk, kl, Cc;
      if (kk < 4){       Wp = Wv1; tb = tp + 64;   nk = 3; kl = kk - 1; Cc = 144; }
      else if (kk < 9){  Wp = Wv2; tb = tp + 496;  nk = 5; kl = kk - 4; Cc = 176; }
      else {             Wp = Wv3; tb = tp + 1376; nk = 7; kl = kk - 9; Cc = 160; }
      float acc = 0.f;
      for (int c = 0; c < Cc; ++c) acc += tb[c*nk + kl] * Wp[c*16 + dch];
      out = acc * rsqrtf((float)Cc);
    }
    Vn[(size_t)e*256 + idx] = out;
  }
}

// ---------------- layer-2 scalar tensor product + xc concat -------------------
__global__ void tp2_kernel(const float* __restrict__ node, const float* __restrict__ Vn,
                           const float* __restrict__ x1, const float* __restrict__ Tg,
                           const int* __restrict__ senders, const float* __restrict__ vareps,
                           float* __restrict__ xc){
  int idx = blockIdx.x*blockDim.x + threadIdx.x;
  if (idx >= EDGES*320) return;
  int e = idx / 320; int j = idx % 320;
  if (j < 256){ xc[(size_t)e*320 + j] = x1[(size_t)e*256 + j]; return; }
  int c = j - 256; int l = c >> 4; int m = c & 15;
  int n = 2*l + 1; int o = SLO[l];
  float ve  = vareps[0];
  float eps = rsqrtf(1.f + ve*ve);
  int s = senders[e];
  const float* Tp = Tg + P_TOFF[l];   // paths 0..3 are (l,l)->0
  float acc = 0.f;
  for (int a = 0; a < n; ++a){
    float av = eps * node[(size_t)s*256 + m*16 + o + a];
    for (int b = 0; b < n; ++b)
      acc += Tp[a*n + b] * av * Vn[(size_t)e*256 + m*16 + o + b];
  }
  xc[(size_t)e*320 + j] = acc;   // scale = sqrt(1)
}

// ---------------- final projection --------------------------------------------
__global__ void out_kernel(const float* __restrict__ xh, const float* __restrict__ Wout,
                           const float* __restrict__ envb, float* __restrict__ out){
  int e = blockIdx.x*blockDim.x + threadIdx.x;
  if (e >= EDGES) return;
  float s = 0.f;
  for (int j = 0; j < 128; ++j) s += xh[(size_t)e*128 + j] * Wout[j];
  out[e] = envb[e] * s * 0.08838834764831845f;   // 1/sqrt(128)
}

// ---------------- launch -------------------------------------------------------
extern "C" void kernel_launch(void* const* d_in, const int* in_sizes, int n_in,
                              void* d_out, int out_size, void* d_ws, size_t ws_size,
                              hipStream_t stream){
  const float* vectors = (const float*)d_in[0];
  const float* vareps  = (const float*)d_in[1];
  const float* alpha   = (const float*)d_in[2];
  const float* emb     = (const float*)d_in[3];
  const float* W_tb1   = (const float*)d_in[4];
  const float* W_tb2   = (const float*)d_in[5];
  const float* W_tb3   = (const float*)d_in[6];
  const float* W_tb4   = (const float*)d_in[7];
  const float* W_w0    = (const float*)d_in[8];
  const float* W_w1    = (const float*)d_in[9];
  const float* W_l11   = (const float*)d_in[10];
  const float* W_l12   = (const float*)d_in[11];
  const float* W_l13   = (const float*)d_in[12];
  const float* W_v1    = (const float*)d_in[13];
  const float* W_v2    = (const float*)d_in[14];
  const float* W_v3    = (const float*)d_in[15];
  const float* W_w2    = (const float*)d_in[16];
  const float* W_l21   = (const float*)d_in[17];
  const float* W_l22   = (const float*)d_in[18];
  const float* W_l23   = (const float*)d_in[19];
  const float* W_h     = (const float*)d_in[20];
  const float* W_out_  = (const float*)d_in[21];
  const int*   senders = (const int*)d_in[22];
  const int*   recv    = (const int*)d_in[23];
  const int*   species = (const int*)d_in[24];
  float* out = (float*)d_out;

  // workspace layout
  float* base  = (float*)d_ws;
  float* Tg    = base;                         // 3456 floats (3436 used)
  _Float16* wh = (_Float16*)(base + 3456);     // f16 weight arena (<= 524288 halves)
  float* envb  = base + 3456 + 262144;         // E
  float* Yb    = envb  + EDGES;                // 16E
  float* w0b   = Yb    + 16*(size_t)EDGES;     // 16E
  float* w1b   = w0b   + 16*(size_t)EDGES;     // 16E (reused for w2)
  float* node  = w1b   + 16*(size_t)EDGES;     // NNODES*256
  float* Vnb   = node  + (size_t)NNODES*256;   // 256E
  float* xcur  = Vnb   + 256*(size_t)EDGES;    // 256E
  float* xnext = xcur  + 256*(size_t)EDGES;    // 256E
  float* bufA  = xnext + 256*(size_t)EDGES;    // 320E (x0 / xc / ping-pong)
  float* bufB  = bufA  + 320*(size_t)EDGES;    // 256E

  const dim3 B256(256);
  const dim3 GEDGE((EDGES + 255)/256);
  const int  nodeN = NNODES*256;

  // 0) coupling tensors + weight conversion (all independent)
  cg_init_kernel<<<dim3(34), dim3(64), 0, stream>>>(Tg);
  size_t woff = 0;
  auto prep = [&](const float* W, int K, int N) -> _Float16* {
    int KP = ((K+31)/32)*32;
    _Float16* dst = wh + woff;
    int n = N*KP;
    wprep_kernel<<<dim3((n+255)/256), B256, 0, stream>>>(W, dst, K, N, KP);
    woff += (size_t)n;
    return dst;
  };
  _Float16* T_tb1 = prep(W_tb1,  72,  32);
  _Float16* T_tb2 = prep(W_tb2,  32,  64);
  _Float16* T_tb3 = prep(W_tb3,  64, 128);
  _Float16* T_tb4 = prep(W_tb4, 128, 256);
  _Float16* T_w0  = prep(W_w0,  256,  16);
  _Float16* T_w1  = prep(W_w1,  256,  16);
  _Float16* T_l11 = prep(W_l11, 320, 256);
  _Float16* T_l12 = prep(W_l12, 256, 256);
  _Float16* T_l13 = prep(W_l13, 256, 256);
  _Float16* T_w2  = prep(W_w2,  256,  16);
  _Float16* T_l21 = prep(W_l21, 320, 256);
  _Float16* T_l22 = prep(W_l22, 256, 256);
  _Float16* T_l23 = prep(W_l23, 256, 256);
  _Float16* T_h   = prep(W_h,   256, 128);

  // 1) edge geometry, bessel, Y, embedding concat -> bufA[E,72]
  edge_init_kernel<<<GEDGE, B256, 0, stream>>>(vectors, emb, senders, recv, species, envb, Yb, bufA);
  // 2) two-body MLP: 72->32->64->128->256, env scale
  gemm_wmma<72,  32, 1><<<dim3(EDGES/(16*rt_for( 72, 32))), B256, 0, stream>>>(bufA, T_tb1, bufB, nullptr, nullptr, nullptr);
  gemm_wmma<32,  64, 1><<<dim3(EDGES/(16*rt_for( 32, 64))), B256, 0, stream>>>(bufB, T_tb2, bufA, nullptr, nullptr, nullptr);
  gemm_wmma<64, 128, 1><<<dim3(EDGES/(16*rt_for( 64,128))), B256, 0, stream>>>(bufA, T_tb3, bufB, nullptr, nullptr, nullptr);
  gemm_wmma<128,256, 2><<<dim3(EDGES/(16*rt_for(128,256))), B256, 0, stream>>>(bufB, T_tb4, xcur, envb, nullptr, nullptr);
  // 3) edge weights w0, w1
  gemm_wmma<256, 16, 0><<<dim3(EDGES/(16*rt_for(256, 16))), B256, 0, stream>>>(xcur, T_w0, w0b, nullptr, nullptr, nullptr);
  gemm_wmma<256, 16, 0><<<dim3(EDGES/(16*rt_for(256, 16))), B256, 0, stream>>>(xcur, T_w1, w1b, nullptr, nullptr, nullptr);
  // 4) segment_sum layer 1
  zero_kernel<<<dim3((nodeN+255)/256), B256, 0, stream>>>(node, nodeN);
  scatter_kernel<<<dim3(EDGES), B256, 0, stream>>>(w1b, Yb, senders, node);
  // 5) tensor product + Vn + xc
  tp1_kernel<<<dim3(EDGES), dim3(128), 0, stream>>>(node, w0b, Yb, xcur, Tg, senders, vareps,
                                                    W_v1, W_v2, W_v3, bufA, Vnb);
  // 6) latent MLP layer 1 + residual -> xnext
  gemm_wmma<320,256, 1><<<dim3(EDGES/(16*rt_for(320,256))), B256, 0, stream>>>(bufA, T_l11, bufB, nullptr, nullptr, nullptr);
  gemm_wmma<256,256, 1><<<dim3(EDGES/(16*rt_for(256,256))), B256, 0, stream>>>(bufB, T_l12, bufA, nullptr, nullptr, nullptr);
  gemm_wmma<256,256, 3><<<dim3(EDGES/(16*rt_for(256,256))), B256, 0, stream>>>(bufA, T_l13, xnext, envb, xcur, alpha);
  // 7) layer 2: w2, segment_sum, scalar TP
  gemm_wmma<256, 16, 0><<<dim3(EDGES/(16*rt_for(256, 16))), B256, 0, stream>>>(xnext, T_w2, w1b, nullptr, nullptr, nullptr);
  zero_kernel<<<dim3((nodeN+255)/256), B256, 0, stream>>>(node, nodeN);
  scatter_kernel<<<dim3(EDGES), B256, 0, stream>>>(w1b, Yb, senders, node);
  tp2_kernel<<<dim3((EDGES*320+255)/256), B256, 0, stream>>>(node, Vnb, xnext, Tg, senders, vareps, bufA);
  // 8) latent MLP layer 2 + residual -> xcur (x2)
  gemm_wmma<320,256, 1><<<dim3(EDGES/(16*rt_for(320,256))), B256, 0, stream>>>(bufA, T_l21, bufB, nullptr, nullptr, nullptr);
  gemm_wmma<256,256, 1><<<dim3(EDGES/(16*rt_for(256,256))), B256, 0, stream>>>(bufB, T_l22, bufA, nullptr, nullptr, nullptr);
  gemm_wmma<256,256, 3><<<dim3(EDGES/(16*rt_for(256,256))), B256, 0, stream>>>(bufA, T_l23, xcur, envb, xnext, alpha);
  // 9) head: 256->128, then 128->1 with env
  gemm_wmma<256,128, 0><<<dim3(EDGES/(16*rt_for(256,128))), B256, 0, stream>>>(xcur, T_h, bufB, nullptr, nullptr, nullptr);
  out_kernel<<<GEDGE, B256, 0, stream>>>(bufB, W_out_, envb, out);
  (void)in_sizes; (void)n_in; (void)out_size; (void)ws_size;
}
